// AttentionHead_18416819765964
// MI455X (gfx1250) — matrix operand order repaired and checked by reading, not compiled, not run on hardware
//
#include <hip/hip_runtime.h>

#define BATCH 4
#define CIN   256
#define DVC   128
#define NPOS  4096   // H*W = 64*64
#define WAVES 4      // waves per attention workgroup

typedef __attribute__((ext_vector_type(16))) __bf16 v16bf;
typedef __attribute__((ext_vector_type(8)))  __bf16 v8bf;
typedef __attribute__((ext_vector_type(8)))  float  v8f;
typedef __attribute__((ext_vector_type(4)))  float  v4f;

union BF16x16 { v16bf v; v8bf h[2]; };

// ---------------------------------------------------------------------------
// WMMA wrapper: D = A(16x32 bf16) * B(32x16 bf16) + C(16x16 f32)
// ---------------------------------------------------------------------------
__device__ __forceinline__ v8f wmma_bf16(v16bf a, v16bf b, v8f c) {
  return __builtin_amdgcn_wmma_f32_16x16x32_bf16(
      /*neg_a=*/false, a, /*neg_b=*/false, b,
      /*c_mod=*/(short)0, c, /*reuse_a=*/false, /*reuse_b=*/false);
}

// A fragment (16x32, M x K), source row-major [rows][ld] bf16.
// Lane L<16 : row row0+L, K = k0+0..7 (v0-3) and k0+16..23 (v4-7)
// Lane L>=16: row row0+L-16, K = k0+8..15 and k0+24..31
__device__ __forceinline__ v16bf load_frag_a(const __bf16* base, int ld,
                                             int row0, int k0, int lane) {
  int r  = row0 + (lane & 15);
  int ko = k0 + ((lane >> 4) << 3);
  const __bf16* p = base + (size_t)r * ld + ko;
  BF16x16 f;
  f.h[0] = *(const v8bf*)(p);
  f.h[1] = *(const v8bf*)(p + 16);
  return f.v;
}

// B fragment (32x16, K x N). Source is the TRANSPOSED operand stored row-major:
// row (col0 + lane&15) supplies two runs of 8 K values per lane (k0+8h, k0+16+8h).
// Lane n<16 : col col0+n, K = k0+0..7 and k0+16..23 ; lane n>=16: col col0+n-16, K = k0+8..15 and k0+24..31
__device__ __forceinline__ v16bf load_frag_b(const __bf16* base, int ld,
                                             int col0, int k0, int lane) {
  int c  = col0 + (lane & 15);
  int ko = k0 + ((lane >> 4) << 3);
  const __bf16* p = base + (size_t)c * ld + ko;
  BF16x16 f;
  f.h[0] = *(const v8bf*)(p);
  f.h[1] = *(const v8bf*)(p + 16);
  return f.v;
}

// ---------------------------------------------------------------------------
// Kernel 1: x [B][C][N] f32  ->  xT [B][N][C] bf16   (LDS-tiled transpose)
// ---------------------------------------------------------------------------
__global__ __launch_bounds__(256) void k_transpose_bf16(
    const float* __restrict__ x, __bf16* __restrict__ xT) {
  __shared__ float tile[32][33];
  int b  = blockIdx.z;
  int c0 = blockIdx.y * 32;
  int n0 = blockIdx.x * 32;
  int tx = threadIdx.x;          // 0..31
  int ty = threadIdx.y;          // 0..7
  const float* xb = x + ((size_t)b * CIN + c0) * NPOS + n0;
#pragma unroll
  for (int i = 0; i < 32; i += 8)
    tile[ty + i][tx] = xb[(size_t)(ty + i) * NPOS + tx];
  __syncthreads();
  __bf16* xo = xT + ((size_t)b * NPOS + n0) * CIN + c0;
#pragma unroll
  for (int i = 0; i < 32; i += 8)
    xo[(size_t)(ty + i) * CIN + tx] = (__bf16)tile[tx][ty + i];
}

// ---------------------------------------------------------------------------
// Kernel 2: Wq/Wk/Wv f32 [VC][C] -> Wb bf16 [3][VC][C]
// ---------------------------------------------------------------------------
__global__ __launch_bounds__(256) void k_convert_w(
    const float* __restrict__ wq, const float* __restrict__ wk,
    const float* __restrict__ wv, __bf16* __restrict__ o) {
  int i = blockIdx.x * 256 + threadIdx.x;
  const int sz = DVC * CIN;
  if (i >= 3 * sz) return;
  const float* src = (i < sz) ? wq : (i < 2 * sz ? wk : wv);
  o[i] = (__bf16)src[i % sz];
}

// ---------------------------------------------------------------------------
// Kernel 3: QKV projection GEMMs (one wave = 16 positions x all 128 d, one mat)
//   mat 0/1 (Q,K): D[M=d][N=pos] = W * x   -> packed b128 stores, pos-major
//   mat 2   (V)  : D[M=pos][N=d] = xT * Wt -> packed b128 stores, d-major
// ---------------------------------------------------------------------------
__global__ __launch_bounds__(32) void k_qkv_proj(
    const __bf16* __restrict__ xT, const __bf16* __restrict__ Wb,
    __bf16* __restrict__ Qp, __bf16* __restrict__ Kp, __bf16* __restrict__ Vd) {
  int lane = threadIdx.x;
  int pt   = blockIdx.x;                       // 0..255 position tile
  int mat  = blockIdx.y;                       // 0=Q 1=K 2=V
  int b    = blockIdx.z;
  const __bf16* Xb = xT + ((size_t)b * NPOS + pt * 16) * CIN;  // [16][CIN]
  const __bf16* W  = Wb + (size_t)mat * (DVC * CIN);           // [VC][CIN]

  v8f acc[8];
#pragma unroll
  for (int dt = 0; dt < 8; ++dt)
#pragma unroll
    for (int r = 0; r < 8; ++r) acc[dt][r] = 0.0f;

  int n  = lane & 15;
  int hi = lane >> 4;

  if (mat < 2) {
    // D[M=d][N=pos]: A = W tile rows, B = xT rows (pos columns)
    for (int ck = 0; ck < CIN; ck += 32) {
      v16bf bf = load_frag_b(Xb, CIN, 0, ck, lane);
#pragma unroll
      for (int dt = 0; dt < 8; ++dt) {
        v16bf af = load_frag_a(W, CIN, dt * 16, ck, lane);
        acc[dt] = wmma_bf16(af, bf, acc[dt]);
      }
    }
    // element (d = dt*16 + hi*8 + r, pos = pt*16 + n) -> Q[pos][d], 8 contiguous d
    __bf16* O = (mat == 0 ? Qp : Kp) + (size_t)b * NPOS * DVC;
#pragma unroll
    for (int dt = 0; dt < 8; ++dt) {
      v8bf pk;
#pragma unroll
      for (int r = 0; r < 8; ++r) pk[r] = (__bf16)acc[dt][r];
      *(v8bf*)(O + (size_t)(pt * 16 + n) * DVC + dt * 16 + hi * 8) = pk;
    }
  } else {
    // D[M=pos][N=d]: A = xT tile, B = W rows (d columns)
    for (int ck = 0; ck < CIN; ck += 32) {
      v16bf af = load_frag_a(Xb, CIN, 0, ck, lane);
#pragma unroll
      for (int dt = 0; dt < 8; ++dt) {
        v16bf bf = load_frag_b(W, CIN, dt * 16, ck, lane);
        acc[dt] = wmma_bf16(af, bf, acc[dt]);
      }
    }
    // element (pos = pt*16 + hi*8 + r, d = dt*16 + n) -> V[d][pos], 8 contiguous pos
    __bf16* O = Vd + (size_t)b * DVC * NPOS;
#pragma unroll
    for (int dt = 0; dt < 8; ++dt) {
      v8bf pk;
#pragma unroll
      for (int r = 0; r < 8; ++r) pk[r] = (__bf16)acc[dt][r];
      *(v8bf*)(O + (size_t)(dt * 16 + n) * NPOS + pt * 16 + hi * 8) = pk;
    }
  }
}

// ---------------------------------------------------------------------------
// Kernel 4: flash attention, 4 waves/WG sharing K/V tiles via double-buffered
// LDS. Each wave owns a 16-query tile; per 32-key chunk: 8 WMMAs for S,
// online softmax, LDS C->A transpose of P, 8 WMMAs for O.
// ---------------------------------------------------------------------------
__global__ __launch_bounds__(WAVES * 32) void k_attn(
    const __bf16* __restrict__ Qp, const __bf16* __restrict__ Kp,
    const __bf16* __restrict__ Vd, float* __restrict__ out) {
  __shared__ __bf16 kbuf[2][32 * DVC];        // 2 x 8KB: [key][d], d contiguous
  __shared__ __bf16 vbuf[2][DVC * 32];        // 2 x 8KB: [d][key], key contiguous
  __shared__ __bf16 pbuf[WAVES][16 * 32];     // per-wave P tile [16 q][32 k]

  int tid  = threadIdx.x;
  int lane = tid & 31;
  int wv   = tid >> 5;
  int qt   = blockIdx.x * WAVES + wv;          // query tile 0..255
  int b    = blockIdx.y;
  const __bf16* Q = Qp + ((size_t)b * NPOS + qt * 16) * DVC;  // [16][DVC]
  const __bf16* K = Kp + (size_t)b * NPOS * DVC;              // [NPOS][DVC]
  const __bf16* V = Vd + (size_t)b * DVC * NPOS;              // [DVC][NPOS]

  // Cooperative stage of one 32-key chunk into LDS buffer `buf`.
  auto stage = [&](int buf, int kb) {
    // K tile: 32*DVC bf16 contiguous in global -> straight copy (8 KB).
    const v8bf* ksrc = (const v8bf*)(K + (size_t)kb * DVC);
#pragma unroll
    for (int i = 0; i < 4; ++i) {
      int idx = tid + i * (WAVES * 32);        // 0..511 v8bf chunks
      *(v8bf*)&kbuf[buf][idx * 8] = ksrc[idx];
    }
    // V tile: row d = tid (DVC == 128 threads), 32 bf16 from V[d][kb..kb+31].
    const __bf16* vsrc = V + (size_t)tid * NPOS + kb;
#pragma unroll
    for (int p = 0; p < 4; ++p)
      *(v8bf*)&vbuf[buf][tid * 32 + p * 8] = *(const v8bf*)(vsrc + p * 8);
  };

  // Q fragments resident: 4 chunks of K-dim 32
  v16bf aq[4];
#pragma unroll
  for (int kc = 0; kc < 4; ++kc) aq[kc] = load_frag_a(Q, DVC, 0, kc * 32, lane);

  v8f   o[8];
  float m[8], l[8];
#pragma unroll
  for (int dt = 0; dt < 8; ++dt)
#pragma unroll
    for (int r = 0; r < 8; ++r) o[dt][r] = 0.0f;
#pragma unroll
  for (int r = 0; r < 8; ++r) { m[r] = -3.0e38f; l[r] = 0.0f; }

  int n  = lane & 15;
  int hi = lane >> 4;

  stage(0, 0);
  __syncthreads();

  for (int c = 0; c < NPOS / 32; ++c) {
    int kb  = c * 32;
    int cur = c & 1;
    // Prefetch next chunk into the other buffer (safe: last read before the previous barrier).
    if (c + 1 < NPOS / 32) stage(cur ^ 1, kb + 32);

    // ---- S = Q K^T for 32 keys (two 16x16 tiles), K frags from LDS ----
    v8f s0, s1;
#pragma unroll
    for (int r = 0; r < 8; ++r) { s0[r] = 0.0f; s1[r] = 0.0f; }
#pragma unroll
    for (int kc = 0; kc < 4; ++kc) {
      s0 = wmma_bf16(aq[kc], load_frag_b(&kbuf[cur][0], DVC, 0,  kc * 32, lane), s0);
      s1 = wmma_bf16(aq[kc], load_frag_b(&kbuf[cur][0], DVC, 16, kc * 32, lane), s1);
    }

    // ---- online softmax (rows: lanes<16 -> M=r, lanes>=16 -> M=8+r) ----
    float scale[8];
#pragma unroll
    for (int r = 0; r < 8; ++r) {
      float mx = fmaxf(s0[r], s1[r]);
      mx = fmaxf(mx, __shfl_xor(mx, 1, 16));
      mx = fmaxf(mx, __shfl_xor(mx, 2, 16));
      mx = fmaxf(mx, __shfl_xor(mx, 4, 16));
      mx = fmaxf(mx, __shfl_xor(mx, 8, 16));
      float mn = fmaxf(m[r], mx);
      scale[r] = __expf(m[r] - mn);
      m[r] = mn;
      float p0 = __expf(s0[r] - mn);
      float p1 = __expf(s1[r] - mn);
      s0[r] = p0; s1[r] = p1;
      float rs = p0 + p1;
      rs += __shfl_xor(rs, 1, 16);
      rs += __shfl_xor(rs, 2, 16);
      rs += __shfl_xor(rs, 4, 16);
      rs += __shfl_xor(rs, 8, 16);
      l[r] = l[r] * scale[r] + rs;
    }
#pragma unroll
    for (int dt = 0; dt < 8; ++dt)
#pragma unroll
      for (int r = 0; r < 8; ++r) o[dt][r] *= scale[r];

    // ---- P: C-layout -> A-layout via wave-private LDS (DS is in-order) ----
#pragma unroll
    for (int r = 0; r < 8; ++r) {
      int row = hi * 8 + r;
      pbuf[wv][row * 32 + n]      = (__bf16)s0[r];
      pbuf[wv][row * 32 + n + 16] = (__bf16)s1[r];
    }
    asm volatile("s_wait_dscnt 0" ::: "memory");
    BF16x16 pa;
    {
      int rr = lane & 15;
      int ko = (lane >> 4) * 8;
      pa.h[0] = *(const v8bf*)&pbuf[wv][rr * 32 + ko];
      pa.h[1] = *(const v8bf*)&pbuf[wv][rr * 32 + ko + 16];
    }

    // ---- O += P * V^T (8 d-tiles), V frags from LDS ----
#pragma unroll
    for (int dt = 0; dt < 8; ++dt)
      o[dt] = wmma_bf16(pa.v, load_frag_b(&vbuf[cur][0], 32, dt * 16, 0, lane), o[dt]);

    __syncthreads();   // next-buffer staging visible; protects buffer reuse
  }

  // ---- epilogue: divide by l, store out[b][d][pos] (8 contiguous pos/lane) ----
  float invl[8];
#pragma unroll
  for (int r = 0; r < 8; ++r) invl[r] = 1.0f / l[r];
#pragma unroll
  for (int dt = 0; dt < 8; ++dt) {
    float* op = out + ((size_t)b * DVC + dt * 16 + n) * NPOS + qt * 16 + hi * 8;
    v4f t0, t1;
#pragma unroll
    for (int r = 0; r < 4; ++r) { t0[r] = o[dt][r] * invl[r]; t1[r] = o[dt][r + 4] * invl[r + 4]; }
    *(v4f*)op       = t0;
    *(v4f*)(op + 4) = t1;
  }
}

// ---------------------------------------------------------------------------
// Launch
// ---------------------------------------------------------------------------
extern "C" void kernel_launch(void* const* d_in, const int* in_sizes, int n_in,
                              void* d_out, int out_size, void* d_ws, size_t ws_size,
                              hipStream_t stream) {
  (void)in_sizes; (void)n_in; (void)out_size; (void)ws_size;
  const float* x  = (const float*)d_in[0];
  const float* Wq = (const float*)d_in[1];
  const float* Wk = (const float*)d_in[2];
  const float* Wv = (const float*)d_in[3];
  float* out = (float*)d_out;

  char* ws = (char*)d_ws;
  __bf16* xT = (__bf16*)(ws);                               // 8 MB  [B][N][C]
  __bf16* Wb = (__bf16*)(ws + (8ull  << 20));               // 192KB [3][VC][C]
  __bf16* Qp = (__bf16*)(ws + (9ull  << 20));               // 4 MB  [B][N][VC]
  __bf16* Kp = (__bf16*)(ws + (13ull << 20));               // 4 MB  [B][N][VC]
  __bf16* Vd = (__bf16*)(ws + (17ull << 20));               // 4 MB  [B][VC][N]

  k_transpose_bf16<<<dim3(NPOS / 32, CIN / 32, BATCH), dim3(32, 8), 0, stream>>>(x, xT);
  k_convert_w<<<(3 * DVC * CIN + 255) / 256, 256, 0, stream>>>(Wq, Wk, Wv, Wb);
  k_qkv_proj<<<dim3(NPOS / 16, 3, BATCH), 32, 0, stream>>>(xT, Wb, Qp, Kp, Vd);
  k_attn<<<dim3(NPOS / (16 * WAVES), BATCH), WAVES * 32, 0, stream>>>(Qp, Kp, Vd, out);
}
